// SelfAttention_51479478010551
// MI455X (gfx1250) — matrix operand order repaired and checked by reading, not compiled, not run on hardware
//
#include <hip/hip_runtime.h>
#include <hip/hip_bf16.h>

#define B_   2
#define S_   2048
#define HID_ 1024
#define NH_  16
#define HD_  64
#define M_   (B_*S_)   // 4096

typedef __attribute__((ext_vector_type(16))) __bf16 bfx16;
typedef __attribute__((ext_vector_type(8)))  float  v8f;
typedef __attribute__((ext_vector_type(4)))  unsigned int tdm_v4u;
typedef __attribute__((ext_vector_type(8)))  int          tdm_v8i;
typedef __attribute__((ext_vector_type(4)))  int          tdm_v4i;

union Frag16 { unsigned short s[16]; uint4 u[2]; bfx16 v; };

__device__ __forceinline__ unsigned short f2bf(float f) {
  unsigned int u = __float_as_uint(f);
  u += 0x7FFFu + ((u >> 16) & 1u);          // round-to-nearest-even
  return (unsigned short)(u >> 16);
}
__device__ __forceinline__ float bf2f(unsigned short s) {
  return __uint_as_float(((unsigned int)s) << 16);
}

#if __has_builtin(__builtin_amdgcn_cvt_pk_bf16_f32)
#pragma message("CVT path: v_cvt_pk_bf16_f32 builtin")
typedef __attribute__((ext_vector_type(2))) __bf16 bfx2;
__device__ __forceinline__ unsigned pk2(float a, float b) {
  union { bfx2 v; unsigned u; } c;
  c.v = __builtin_amdgcn_cvt_pk_bf16_f32(a, b);
  return c.u;
}
#else
#pragma message("CVT path: manual RNE bf16 pack")
__device__ __forceinline__ unsigned pk2(float a, float b) {
  return (unsigned)f2bf(a) | ((unsigned)f2bf(b) << 16);
}
#endif

__device__ __forceinline__ uint4 pack8(const float4 a, const float4 b) {
  uint4 o;
  o.x = pk2(a.x, a.y);
  o.y = pk2(a.z, a.w);
  o.z = pk2(b.x, b.y);
  o.w = pk2(b.z, b.w);
  return o;
}

#define WMMA_BF16(A, Bf, C) \
  __builtin_amdgcn_wmma_f32_16x16x32_bf16(false, (A), false, (Bf), (short)0, (C), false, false)

// The TDM writes LDS behind LLVM's back. A plain memory clobber is not enough:
// the shared arrays are non-escaping, so alias analysis proves the asm cannot
// touch them and folds the (apparently never-stored) LDS loads to undef.
// Passing the LDS base pointer INTO the asm makes it escape -> the asm is a
// potential writer -> subsequent fragment loads must be honored.
__device__ __forceinline__ void tdm_fence(void* lds_base) {
  asm volatile("" : "+v"(lds_base) :: "memory");
}

// ---------------- Tensor Data Mover helper -----------------------------------
#if __has_builtin(__builtin_amdgcn_tensor_load_to_lds)
#define USE_TDM 1
#pragma message("TDM path: tensor_load_to_lds, double-buffered")
#if defined(__clang_major__) && __clang_major__ >= 23
#define TDM_6ARG 1
#pragma message("TDM arity: 6-arg (clang>=23)")
#else
#define TDM_6ARG 0
#pragma message("TDM arity: 5-arg (clang<23)")
#endif
#else
#define USE_TDM 0
#pragma message("TDM path: NOT available, manual staging")
#endif

#if USE_TDM
// 2D bf16 tile -> LDS. tile_w = contiguous elements/row, tile_h = rows,
// stride = row pitch in elements. LDS pad after every 2^(padInt+1) DWORDs of
// (padAmt+1) DWORDs lets TDM write our bank-padded LDS tiles directly.
__device__ __forceinline__ void tdm_load_2d(unsigned ldsOff, const unsigned short* g,
                                            unsigned tile_w, unsigned tile_h,
                                            unsigned stride, unsigned padInt, unsigned padAmt) {
  const unsigned long long ga = (unsigned long long)(size_t)g;
  tdm_v4u g0 = { 1u,                      // count=1, user descriptor
                 ldsOff,                  // LDS byte address
                 (unsigned)(ga & 0xFFFFFFFFu),
                 (unsigned)((ga >> 32) & 0x1FFFFFFu) | 0x80000000u };   // type=2
  tdm_v8i g1 = { (int)((1u << 16) | (1u << 20) | (padInt << 22) | (padAmt << 25)),
                 // ^ data_size=2B | pad_enable | pad_interval | pad_amount
                 (int)(tile_w << 16),     // tensor_dim0
                 (int)(tile_h << 16),     // tensor_dim1
                 (int)(tile_w << 16),     // tile_dim0
                 (int)tile_h,             // tile_dim1 (tile_dim2=0)
                 (int)stride,             // tensor_dim0_stride
                 0, 0 };
  tdm_v4i z4 = { 0, 0, 0, 0 };
#if TDM_6ARG
  tdm_v8i z8 = { 0, 0, 0, 0, 0, 0, 0, 0 };
  __builtin_amdgcn_tensor_load_to_lds(g0, g1, z4, z4, z8, 0);
#else
  __builtin_amdgcn_tensor_load_to_lds(g0, g1, z4, z4, 0);
#endif
}
#endif

// ---------------- f32 -> bf16 conversion of X (once, reused by 3 GEMMs) ------
__global__ __launch_bounds__(256) void conv_x(const float* __restrict__ X,
                                              unsigned short* __restrict__ Xb) {
  const size_t i = ((size_t)blockIdx.x * 256 + threadIdx.x) * 8;
  float4 a = *(const float4*)(X + i), b = *(const float4*)(X + i + 4);
  *(uint4*)(Xb + i) = pack8(a, b);
}

// ---------------- W[k][n] f32 -> WT[n][k] bf16 (LDS-tiled transpose) ---------
__global__ __launch_bounds__(256) void conv_wT(const float* __restrict__ W,
                                               unsigned short* __restrict__ WT) {
  __shared__ float tile[32][33];
  const int tx = threadIdx.x & 31, ty = threadIdx.x >> 5;   // ty 0..7
  const int k0 = blockIdx.x * 32, n0 = blockIdx.y * 32;
#pragma unroll
  for (int i = 0; i < 4; ++i)
    tile[ty + i * 8][tx] = W[(size_t)(k0 + ty + i * 8) * HID_ + n0 + tx];
  __syncthreads();
#pragma unroll
  for (int i = 0; i < 4; ++i)
    WT[(size_t)(n0 + ty + i * 8) * HID_ + k0 + tx] = f2bf(tile[tx][ty + i * 8]);
}

// ---------------- QKV projection: C = Xb @ WT^T + bias (all bf16 tiles) ------
// 64x64 block tile, BK=32, ping-pong A/B tiles fed by TDM (or vector staging).
#define GA_WORDS 2560                 // 64 rows * 40-half padded stride
#define GA_BYTES (GA_WORDS * 2)
__global__ __launch_bounds__(256) void qkv_gemm(const unsigned short* __restrict__ Xb,
                                                const unsigned short* __restrict__ WT,
                                                const float* __restrict__ bias,
                                                unsigned short* __restrict__ out) {
  __shared__ __align__(16) unsigned short gsm[4 * GA_WORDS];  // A0 A1 B0 B1
  const int t  = threadIdx.x;
  const int m0 = blockIdx.x * 64, n0 = blockIdx.y * 64;
  const int lane = t & 31, wid = t >> 5;
  const int wr = wid >> 1, wc = wid & 1;
  const int lrow = lane & 15, lhalf = lane >> 4;
  v8f acc0 = {}, acc1 = {};

#if USE_TDM
  if (wid == 0) {
    tdm_load_2d(0 * GA_BYTES, Xb + (size_t)m0 * HID_, 32, 64, HID_, 3, 3);
    tdm_load_2d(2 * GA_BYTES, WT + (size_t)n0 * HID_, 32, 64, HID_, 3, 3);
  }
#endif
  for (int k0 = 0; k0 < HID_; k0 += 32) {
    const int cur = (k0 >> 5) & 1;
#if USE_TDM
    __builtin_amdgcn_s_wait_tensorcnt(0);    // SOPP, EXEC-independent; 0 for non-issuing waves
    __syncthreads();
    tdm_fence(gsm);                          // escape + clobber: keep LDS reads alive
    if (wid == 0 && k0 + 32 < HID_) {        // overlap next tile's DMA with WMMA
      const int nxt = cur ^ 1;
      tdm_load_2d(nxt * GA_BYTES,       Xb + (size_t)m0 * HID_ + k0 + 32, 32, 64, HID_, 3, 3);
      tdm_load_2d((2 + nxt) * GA_BYTES, WT + (size_t)n0 * HID_ + k0 + 32, 32, 64, HID_, 3, 3);
    }
#else
    __syncthreads();
    { const int row = t >> 2, seg = (t & 3) * 8;   // 8 halves/thread, both tiles
      unsigned short (*A)[40] = (unsigned short (*)[40])(gsm + cur * GA_WORDS);
      unsigned short (*Bt)[40] = (unsigned short (*)[40])(gsm + (2 + cur) * GA_WORDS);
      *(uint4*)&A[row][seg]  = *(const uint4*)(Xb + (size_t)(m0 + row) * HID_ + k0 + seg);
      *(uint4*)&Bt[row][seg] = *(const uint4*)(WT + (size_t)(n0 + row) * HID_ + k0 + seg);
    }
    __syncthreads();
#endif
    unsigned short (*A)[40]  = (unsigned short (*)[40])(gsm + cur * GA_WORDS);
    unsigned short (*Bt)[40] = (unsigned short (*)[40])(gsm + (2 + cur) * GA_WORDS);
    Frag16 af, b0, b1;
    af.u[0] = *(const uint4*)&A[wr * 16 + lrow][lhalf * 16];
    af.u[1] = *(const uint4*)&A[wr * 16 + lrow][lhalf * 16 + 8];
    const int kb2 = lhalf * 16;
    b0.u[0] = *(const uint4*)&Bt[wc * 32 + lrow][kb2];
    b0.u[1] = *(const uint4*)&Bt[wc * 32 + lrow][kb2 + 8];
    b1.u[0] = *(const uint4*)&Bt[wc * 32 + 16 + lrow][kb2];
    b1.u[1] = *(const uint4*)&Bt[wc * 32 + 16 + lrow][kb2 + 8];
    acc0 = WMMA_BF16(af.v, b0.v, acc0);
    acc1 = WMMA_BF16(af.v, b1.v, acc1);
  }

  const int rowbase = m0 + wr * 16 + lhalf * 8;
  const float bv0 = bias[n0 + wc * 32 + lrow];
  const float bv1 = bias[n0 + wc * 32 + 16 + lrow];
#pragma unroll
  for (int r = 0; r < 8; ++r) {
    out[(size_t)(rowbase + r) * HID_ + n0 + wc * 32 + lrow]      = f2bf(acc0[r] + bv0);
    out[(size_t)(rowbase + r) * HID_ + n0 + wc * 32 + 16 + lrow] = f2bf(acc1[r] + bv1);
  }
}

// ---------------- RoPE + head split ------------------------------------------
// q,k -> [B,NH,S,HD]; v -> [B,NH,HD,S] (d-major) so the attention V^T tile is a
// plain 2D tile the TDM can fetch without any in-kernel transpose.
__global__ __launch_bounds__(256) void rope_split(const unsigned short* __restrict__ qraw,
                                                  const unsigned short* __restrict__ kraw,
                                                  const unsigned short* __restrict__ vraw,
                                                  unsigned short* __restrict__ q,
                                                  unsigned short* __restrict__ k,
                                                  unsigned short* __restrict__ vt) {
  const int idx = blockIdx.x * 256 + threadIdx.x;       // B*NH*S*32 = 2^21 threads
  const int d = idx & 31;
  const int s = (idx >> 5) & (S_ - 1);
  const int h = (idx >> 16) & (NH_ - 1);
  const int b = idx >> 20;
  if (b >= B_) return;
  const size_t rowIn  = (size_t)(b * S_ + s) * HID_ + h * HD_;
  const size_t rowOut = ((size_t)(b * NH_ + h) * S_ + s) * HD_;
  const size_t vBase  = ((size_t)(b * NH_ + h) * HD_) * S_ + s;  // + d*S_
  const float LN1E4_32 = 0.2878231366242557f;           // ln(10000)/32
  const float ang = (float)s * __expf(-(float)d * LN1E4_32);
  float sn, cs; __sincosf(ang, &sn, &cs);
  { // q: pre-scale by HD^-0.5 = 0.125
    float x1 = bf2f(qraw[rowIn + d]) * 0.125f;
    float x2 = bf2f(qraw[rowIn + d + 32]) * 0.125f;
    q[rowOut + d]      = f2bf(x1 * cs - x2 * sn);
    q[rowOut + d + 32] = f2bf(x2 * cs + x1 * sn);
  }
  {
    float x1 = bf2f(kraw[rowIn + d]);
    float x2 = bf2f(kraw[rowIn + d + 32]);
    k[rowOut + d]      = f2bf(x1 * cs - x2 * sn);
    k[rowOut + d + 32] = f2bf(x2 * cs + x1 * sn);
  }
  vt[vBase + (size_t)d * S_]        = vraw[rowIn + d];
  vt[vBase + (size_t)(d + 32) * S_] = vraw[rowIn + d + 32];
}

// ---------------- Flash attention: per (b,h), 128 q-rows per block -----------
// smem: K ping-pong [kv][d], V^T ping-pong [d][kv] (both pad-72), P per wave.
#define KB_WORDS (64 * 72)            // 4608
#define KB_BYTES (KB_WORDS * 2)
#define VT_OFF   (2 * KB_WORDS)
#define P_OFF    (4 * KB_WORDS)
#define SMEM_WORDS (4 * KB_WORDS + 8 * 16 * 72)

__global__ __launch_bounds__(256) void attn(const unsigned short* __restrict__ Q,
                                            const unsigned short* __restrict__ K,
                                            const unsigned short* __restrict__ Vt,
                                            const float* __restrict__ mask,
                                            float* __restrict__ out) {
  __shared__ __align__(16) unsigned short smem[SMEM_WORDS];
  unsigned short (*Plds)[16][72] = (unsigned short (*)[16][72])(&smem[P_OFF]);

  const int t = threadIdx.x, lane = t & 31, wid = t >> 5;
  const int lrow = lane & 15, lhalf = lane >> 4;
  const int bh = blockIdx.y, b = bh >> 4, h = bh & 15;
  const int q0 = blockIdx.x * 128 + wid * 16;
  const unsigned short* Qh  = Q  + (size_t)bh * S_ * HD_;
  const unsigned short* Kh  = K  + (size_t)bh * S_ * HD_;
  const unsigned short* Vth = Vt + (size_t)bh * HD_ * S_;

  Frag16 qf[2];
#pragma unroll
  for (int c = 0; c < 2; ++c) {
    const unsigned short* p = Qh + (size_t)(q0 + lrow) * HD_ + c * 32 + lhalf * 16;
    qf[c].u[0] = *(const uint4*)p;
    qf[c].u[1] = *(const uint4*)(p + 8);
  }
  v8f acc[4] = {{}, {}, {}, {}};
  float m_i[8], l_i[8];
#pragma unroll
  for (int r = 0; r < 8; ++r) { m_i[r] = -1e30f; l_i[r] = 0.f; }

#if USE_TDM
  if (wid == 0) {
    tdm_load_2d(0,            Kh,  64, 64, HD_, 4, 3);
    tdm_load_2d(VT_OFF * 2,   Vth, 64, 64, S_,  4, 3);
  }
#else
  const int srow = t >> 2, sseg = (t & 3) * 16;   // 64 rows x 4 segs of 16
#endif

  for (int kv0 = 0; kv0 < S_; kv0 += 64) {
    const int cur = (kv0 >> 6) & 1;
#if USE_TDM
    __builtin_amdgcn_s_wait_tensorcnt(0);    // EXEC-independent; 0 for non-issuing waves
    __syncthreads();
    tdm_fence(smem);                         // escape + clobber: keep LDS reads alive
    if (wid == 0 && kv0 + 64 < S_) {         // overlap next tile's DMA
      const int nxt = cur ^ 1;
      tdm_load_2d(nxt * KB_BYTES,              Kh  + (size_t)(kv0 + 64) * HD_, 64, 64, HD_, 4, 3);
      tdm_load_2d(VT_OFF * 2 + nxt * KB_BYTES, Vth + kv0 + 64,                 64, 64, S_,  4, 3);
    }
#else
    __syncthreads();
    { const unsigned short* p = Kh + (size_t)(kv0 + srow) * HD_ + sseg;
      unsigned short (*Kl)[72] = (unsigned short (*)[72])(&smem[cur * KB_WORDS]);
      *(uint4*)&Kl[srow][sseg]     = *(const uint4*)p;
      *(uint4*)&Kl[srow][sseg + 8] = *(const uint4*)(p + 8);
    }
    { const unsigned short* p = Vth + (size_t)srow * S_ + kv0 + sseg;
      unsigned short (*Vl)[72] = (unsigned short (*)[72])(&smem[VT_OFF + cur * KB_WORDS]);
      *(uint4*)&Vl[srow][sseg]     = *(const uint4*)p;
      *(uint4*)&Vl[srow][sseg + 8] = *(const uint4*)(p + 8);
    }
    __syncthreads();
#endif
    unsigned short (*Klds)[72]  = (unsigned short (*)[72])(&smem[cur * KB_WORDS]);
    unsigned short (*VldsT)[72] = (unsigned short (*)[72])(&smem[VT_OFF + cur * KB_WORDS]);

    // scores: 4 kv sub-tiles x 2 K-chunks = 8 WMMA
    v8f sc[4];
#pragma unroll
    for (int nt = 0; nt < 4; ++nt) {
      v8f s = {};
#pragma unroll
      for (int c = 0; c < 2; ++c) {
        Frag16 kf;
        const unsigned short* p = &Klds[nt * 16 + lrow][c * 32 + lhalf * 16];
        kf.u[0] = *(const uint4*)p;
        kf.u[1] = *(const uint4*)(p + 8);
        s = WMMA_BF16(qf[c].v, kf.v, s);
      }
      const float mv = mask[(size_t)b * S_ + kv0 + nt * 16 + lrow];
#pragma unroll
      for (int r = 0; r < 8; ++r) s[r] += mv;
      sc[nt] = s;
    }

    // online softmax (lane holds one column of 8 rows -> reduce across 16 lanes)
    float alpha[8], rs[8];
#pragma unroll
    for (int r = 0; r < 8; ++r) {
      float cm = fmaxf(fmaxf(sc[0][r], sc[1][r]), fmaxf(sc[2][r], sc[3][r]));
#pragma unroll
      for (int off = 1; off < 16; off <<= 1)
        cm = fmaxf(cm, __shfl_xor(cm, off, 32));
      const float mn = fmaxf(m_i[r], cm);
      alpha[r] = __expf(m_i[r] - mn);
      m_i[r] = mn;
      rs[r] = 0.f;
    }
#pragma unroll
    for (int nt = 0; nt < 4; ++nt)
#pragma unroll
      for (int r = 0; r < 8; ++r) {
        const float pe = __expf(sc[nt][r] - m_i[r]);
        sc[nt][r] = pe;
        rs[r] += pe;
      }
#pragma unroll
    for (int r = 0; r < 8; ++r) {
#pragma unroll
      for (int off = 1; off < 16; off <<= 1)
        rs[r] += __shfl_xor(rs[r], off, 32);
      l_i[r] = l_i[r] * alpha[r] + rs[r];
#pragma unroll
      for (int dt = 0; dt < 4; ++dt) acc[dt][r] *= alpha[r];
    }

    // C-layout -> A-layout for P via per-wave LDS tile
#pragma unroll
    for (int nt = 0; nt < 4; ++nt)
#pragma unroll
      for (int r = 0; r < 8; ++r)
        Plds[wid][lhalf * 8 + r][nt * 16 + lrow] = f2bf(sc[nt][r]);
    __syncthreads();

    // P @ V: 2 kv-chunks x 4 d-tiles = 8 WMMA
#pragma unroll
    for (int c = 0; c < 2; ++c) {
      Frag16 pf;
      const unsigned short* pp = &Plds[wid][lrow][c * 32 + lhalf * 16];
      pf.u[0] = *(const uint4*)pp;
      pf.u[1] = *(const uint4*)(pp + 8);
#pragma unroll
      for (int dt = 0; dt < 4; ++dt) {
        Frag16 vf;
        const unsigned short* vp = &VldsT[dt * 16 + lrow][c * 32 + lhalf * 16];
        vf.u[0] = *(const uint4*)vp;
        vf.u[1] = *(const uint4*)(vp + 8);
        acc[dt] = WMMA_BF16(pf.v, vf.v, acc[dt]);
      }
    }
  }

  // normalize and store f32 to [B,S,HID]
#pragma unroll
  for (int dt = 0; dt < 4; ++dt) {
    const int d = h * HD_ + dt * 16 + lrow;
#pragma unroll
    for (int r = 0; r < 8; ++r) {
      const int row = q0 + lhalf * 8 + r;
      out[((size_t)b * S_ + row) * HID_ + d] = acc[dt][r] / l_i[r];
    }
  }
}

// ---------------- launcher ---------------------------------------------------
extern "C" void kernel_launch(void* const* d_in, const int* in_sizes, int n_in,
                              void* d_out, int out_size, void* d_ws, size_t ws_size,
                              hipStream_t stream) {
  const float* hidden = (const float*)d_in[0];
  const float* mask   = (const float*)d_in[1];
  const float* Wq = (const float*)d_in[2];
  const float* bq = (const float*)d_in[3];
  const float* Wk = (const float*)d_in[4];
  const float* bk = (const float*)d_in[5];
  const float* Wv = (const float*)d_in[6];
  const float* bv = (const float*)d_in[7];
  float* out = (float*)d_out;

  const size_t N  = (size_t)M_ * HID_;                 // 4M
  const size_t NW = (size_t)HID_ * HID_;               // 1M
  unsigned short* Xb   = (unsigned short*)d_ws;        // 62 MB total workspace
  unsigned short* WTq  = Xb + N;
  unsigned short* WTk  = WTq + NW;
  unsigned short* WTv  = WTk + NW;
  unsigned short* qraw = WTv + NW;
  unsigned short* kraw = qraw + N;
  unsigned short* vraw = kraw + N;
  unsigned short* qr   = vraw + N;
  unsigned short* kr   = qr + N;
  unsigned short* vt   = kr + N;

  conv_x<<<(int)(N / 8 / 256), 256, 0, stream>>>(hidden, Xb);
  dim3 gt(HID_ / 32, HID_ / 32);                       // (32, 32)
  conv_wT<<<gt, 256, 0, stream>>>(Wq, WTq);
  conv_wT<<<gt, 256, 0, stream>>>(Wk, WTk);
  conv_wT<<<gt, 256, 0, stream>>>(Wv, WTv);

  dim3 gg(M_ / 64, HID_ / 64);                         // (64, 16)
  qkv_gemm<<<gg, 256, 0, stream>>>(Xb, WTq, bq, qraw);
  qkv_gemm<<<gg, 256, 0, stream>>>(Xb, WTk, bk, kraw);
  qkv_gemm<<<gg, 256, 0, stream>>>(Xb, WTv, bv, vraw);

  rope_split<<<(B_ * NH_ * S_ * 32) / 256, 256, 0, stream>>>(qraw, kraw, vraw, qr, kr, vt);
  attn<<<dim3(S_ / 128, B_ * NH_), 256, 0, stream>>>(qr, kr, vt, mask, out);
}
